// CFBlock_86861418594990
// MI455X (gfx1250) — matrix operand order repaired and checked by reading, not compiled, not run on hardware
//
#include <hip/hip_runtime.h>
#include <math.h>

// ---------------------------------------------------------------------------
// CDNA5 (gfx1250) fused SchNet CFConv block, bf16 WMMA + fp32 accumulate.
// Register-blocked: each wave computes 1 row-tile x 4 col-tiles, reusing the
// A fragment across 4 v_wmma_f32_16x16x32_bf16 issues per k-step.
// ---------------------------------------------------------------------------

#define N_NODES  10000
#define N_EDGES  320000
#define D_MODEL  256
#define D_RADIAL 128
#define D_HIDDEN 256
#define D_FF     1024

typedef __attribute__((ext_vector_type(16))) __bf16 v16bf;
typedef __attribute__((ext_vector_type(8)))  __bf16 v8bf;
typedef __attribute__((ext_vector_type(8)))  float  v8f;

union AFrag { v16bf v; v8bf h[2]; };

static __device__ inline float bf2f(__bf16 x) { return (float)x; }

// A-fragment loader (16x32 bf16, M x K): per ISA 7.12.2, lane m=lane&15,
// half=lane>>4 holds K = [half*8, half*8+8) and [16+half*8, 16+half*8+8).
static __device__ inline v16bf load_a_bf16(const __bf16* A, int row, int ldk,
                                           int kbase, int half) {
  AFrag f;
  const __bf16* p = A + (size_t)row * ldk + kbase + half * 8;
  f.h[0] = *(const v8bf*)(p);
  f.h[1] = *(const v8bf*)(p + 16);
  return f.v;
}

// Same but source is fp32 (tf32-style on-the-fly downconvert).
// NT=true marks the load non-temporal (streamed once; keep L2 for hot data).
template <bool NT>
static __device__ inline v16bf load_a_f32(const float* A, int row, int ldk,
                                          int kbase, int half) {
  const float* p = A + (size_t)row * ldk + kbase + half * 8;
  v8f x0, x1;
  if (NT) {
    x0 = __builtin_nontemporal_load((const v8f*)(p));
    x1 = __builtin_nontemporal_load((const v8f*)(p + 16));
  } else {
    x0 = *(const v8f*)(p);
    x1 = *(const v8f*)(p + 16);
  }
  v16bf a;
#pragma unroll
  for (int i = 0; i < 8; ++i) {
    a[i]     = (__bf16)x0[i];
    a[i + 8] = (__bf16)x1[i];
  }
  return a;
}

// ---------------------------------------------------------------------------
// Weight pre-pack: row-major fp32 W[K,N] -> bf16 B-fragment layout.
// B (32x16, K x N): lane l: n = l&15, h = l>>4 holds K [h*8, h*8+8) then [16+h*8, 16+h*8+8).
// Packed: Wp[((ktile*Ntiles + ntile)*32 + lane)*16 + i]
// ---------------------------------------------------------------------------
__global__ __launch_bounds__(256) void pack_w_kernel(const float* __restrict__ W,
                                                     __bf16* __restrict__ Wp,
                                                     int K, int N) {
  int t = blockIdx.x * 256 + threadIdx.x;
  if (t >= K * N) return;
  int k = t / N, n = t - k * N;
  int ktile = k >> 5, koff = k & 31, hi = (koff >> 3) & 1, i = (koff & 7) | ((koff >> 4) << 3);
  int ntile = n >> 4, nn = n & 15;
  size_t dst = ((((size_t)(ktile * (N >> 4) + ntile) << 5) + (hi << 4) + nn) << 4) + i;
  Wp[dst] = (__bf16)W[t];
}

__global__ __launch_bounds__(256) void zero_kernel(float* __restrict__ p, int n) {
  int t = blockIdx.x * 256 + threadIdx.x;
  if (t < n) p[t] = 0.0f;
}

// ---------------------------------------------------------------------------
// LayerNorm over D_MODEL=256, one row per 256-thread block, bf16 output.
// ---------------------------------------------------------------------------
__global__ __launch_bounds__(256) void ln_kernel(const float* __restrict__ x,
                                                 const float* __restrict__ s,
                                                 const float* __restrict__ b,
                                                 __bf16* __restrict__ out) {
  __shared__ float ss[256];
  __shared__ float sq[256];
  int row = blockIdx.x, t = threadIdx.x;
  float v = x[(size_t)row * D_MODEL + t];
  ss[t] = v;
  sq[t] = v * v;
  __syncthreads();
#pragma unroll
  for (int o = 128; o > 0; o >>= 1) {
    if (t < o) { ss[t] += ss[t + o]; sq[t] += sq[t + o]; }
    __syncthreads();
  }
  float mu  = ss[0] * (1.0f / D_MODEL);
  float var = sq[0] * (1.0f / D_MODEL) - mu * mu;
  float r   = rsqrtf(var + 1e-5f);
  out[(size_t)row * D_MODEL + t] = (__bf16)((v - mu) * r * s[t] + b[t]);
}

// ---------------------------------------------------------------------------
// Generic WMMA GEMM: out[M,N] = epilogue(A[M,K] @ Wp + bias, res).
// Block = 128 threads (4 waves); wave w owns 4 consecutive col tiles
// (64 columns), so each block covers 256 output columns per row tile.
// EPI: 0 = bf16 store; 1 = silu(+res) f32; 2 = silu bf16; 3 = +res f32.
// ---------------------------------------------------------------------------
template <int EPI, bool AF32>
__global__ __launch_bounds__(128) void gemm_kernel(const void* __restrict__ Av,
                                                   const __bf16* __restrict__ Bp,
                                                   const float* __restrict__ bias,
                                                   const float* __restrict__ res,
                                                   void* __restrict__ out,
                                                   int K, int N) {
  const int lane = threadIdx.x & 31;
  const int wave = threadIdx.x >> 5;
  const int rt   = blockIdx.x;                        // row tile (16 rows)
  const int ct0  = (blockIdx.y * 4 + wave) * 4;       // first of 4 col tiles
  const int m    = lane & 15;
  const int half = lane >> 4;
  const int Ntiles = N >> 4;
  const int Ktiles = K >> 5;

  v8f c[4];
#pragma unroll
  for (int j = 0; j < 4; ++j) c[j] = (v8f){0.f, 0.f, 0.f, 0.f, 0.f, 0.f, 0.f, 0.f};

  const int arow = rt * 16 + m;
  for (int kt = 0; kt < Ktiles; ++kt) {
    v16bf a = AF32 ? load_a_f32<false>((const float*)Av, arow, K, kt * 32, half)
                   : load_a_bf16((const __bf16*)Av, arow, K, kt * 32, half);
#pragma unroll
    for (int j = 0; j < 4; ++j) {
      v16bf b = *(const v16bf*)(Bp + (((size_t)(kt * Ntiles + ct0 + j)) * 32 + lane) * 16);
      c[j] = __builtin_amdgcn_wmma_f32_16x16x32_bf16(false, a, false, b,
                                                     (short)0, c[j], false, false);
    }
  }

#pragma unroll
  for (int j = 0; j < 4; ++j) {
    const int col = (ct0 + j) * 16 + m;
    const float bv = bias[col];
#pragma unroll
    for (int r = 0; r < 8; ++r) {
      int orow = rt * 16 + r + half * 8;
      size_t oi = (size_t)orow * N + col;
      float v = c[j][r] + bv;
      if (EPI == 0) {
        ((__bf16*)out)[oi] = (__bf16)v;
      } else if (EPI == 1) {
        v = v / (1.0f + expf(-v));
        ((float*)out)[oi] = v + res[oi];
      } else if (EPI == 2) {
        v = v / (1.0f + expf(-v));
        ((__bf16*)out)[oi] = (__bf16)v;
      } else {
        ((float*)out)[oi] = v + res[oi];
      }
    }
  }
}

// ---------------------------------------------------------------------------
// Fused edge pipeline: filt-tile = radial_tile @ W_filt (WMMA), then straight
// from the C fragments: gather h_pre[src] (bf16), multiply, atomic-scatter
// into agg[dst]. filt (327 MB) is never materialized; radial is read NT.
// grid = 20000 edge tiles; block = 128 (4 waves x 4 col tiles = 256 cols).
// ---------------------------------------------------------------------------
__global__ __launch_bounds__(128) void edge_kernel(const float* __restrict__ radial,
                                                   const __bf16* __restrict__ Wp,
                                                   const float* __restrict__ bfilt,
                                                   const __bf16* __restrict__ hpre,
                                                   const int* __restrict__ src,
                                                   const int* __restrict__ dst,
                                                   float* __restrict__ agg) {
  const int lane = threadIdx.x & 31;
  const int wave = threadIdx.x >> 5;
  const int et   = blockIdx.x;            // 16-edge tile
  const int ct0  = wave * 4;              // first of 4 col tiles of D_HIDDEN
  const int m    = lane & 15;
  const int half = lane >> 4;

  v8f c[4];
#pragma unroll
  for (int j = 0; j < 4; ++j) c[j] = (v8f){0.f, 0.f, 0.f, 0.f, 0.f, 0.f, 0.f, 0.f};

  const int arow = et * 16 + m;
#pragma unroll
  for (int kt = 0; kt < D_RADIAL / 32; ++kt) {
    v16bf a = load_a_f32<true>(radial, arow, D_RADIAL, kt * 32, half);
#pragma unroll
    for (int j = 0; j < 4; ++j) {
      v16bf b = *(const v16bf*)(Wp + (((size_t)(kt * (D_HIDDEN / 16) + ct0 + j)) * 32 + lane) * 16);
      c[j] = __builtin_amdgcn_wmma_f32_16x16x32_bf16(false, a, false, b,
                                                     (short)0, c[j], false, false);
    }
  }

  float bv[4];
#pragma unroll
  for (int j = 0; j < 4; ++j) bv[j] = bfilt[(ct0 + j) * 16 + m];

#pragma unroll
  for (int r = 0; r < 8; ++r) {
    int e = et * 16 + r + half * 8;
    int se = src[e];
    int de = dst[e];
#pragma unroll
    for (int j = 0; j < 4; ++j) {
      const int col = (ct0 + j) * 16 + m;
      float hv = bf2f(hpre[(size_t)se * D_HIDDEN + col]);
      float mv = (c[j][r] + bv[j]) * hv;
      unsafeAtomicAdd(&agg[(size_t)de * D_HIDDEN + col], mv);
    }
  }
}

// ---------------------------------------------------------------------------
// Host-side launcher.
// ---------------------------------------------------------------------------
extern "C" void kernel_launch(void* const* d_in, const int* in_sizes, int n_in,
                              void* d_out, int out_size, void* d_ws, size_t ws_size,
                              hipStream_t stream) {
  const float* x      = (const float*)d_in[0];
  const float* radial = (const float*)d_in[1];
  const int*   src    = (const int*)d_in[2];
  const int*   dst    = (const int*)d_in[3];
  const float* ln1_s  = (const float*)d_in[4];
  const float* ln1_b  = (const float*)d_in[5];
  const float* W_pre  = (const float*)d_in[6];
  const float* b_pre  = (const float*)d_in[7];
  const float* W_filt = (const float*)d_in[8];
  const float* b_filt = (const float*)d_in[9];
  const float* W_post = (const float*)d_in[10];
  const float* b_post = (const float*)d_in[11];
  const float* ln2_s  = (const float*)d_in[12];
  const float* ln2_b  = (const float*)d_in[13];
  const float* W_ff1  = (const float*)d_in[14];
  const float* b_ff1  = (const float*)d_in[15];
  const float* W_ff2  = (const float*)d_in[16];
  const float* b_ff2  = (const float*)d_in[17];

  char* ws = (char*)d_ws;
  size_t off = 0;
  auto take = [&](size_t bytes) -> void* {
    void* p = ws + off;
    off += (bytes + 255) & ~(size_t)255;
    return p;
  };

  __bf16* ln1bf = (__bf16*)take((size_t)N_NODES * D_MODEL * 2);
  __bf16* hpre  = (__bf16*)take((size_t)N_NODES * D_HIDDEN * 2);
  float*  agg   = (float*)take((size_t)N_NODES * D_HIDDEN * 4);
  float*  x2    = (float*)take((size_t)N_NODES * D_MODEL * 4);
  __bf16* ln2bf = (__bf16*)take((size_t)N_NODES * D_MODEL * 2);
  __bf16* gbf   = (__bf16*)take((size_t)N_NODES * D_FF * 2);
  __bf16* WpPre  = (__bf16*)take((size_t)D_MODEL * D_HIDDEN * 2);
  __bf16* WpFilt = (__bf16*)take((size_t)D_RADIAL * D_HIDDEN * 2);
  __bf16* WpPost = (__bf16*)take((size_t)D_HIDDEN * D_MODEL * 2);
  __bf16* WpFF1  = (__bf16*)take((size_t)D_MODEL * D_FF * 2);
  __bf16* WpFF2  = (__bf16*)take((size_t)D_FF * D_MODEL * 2);

  // --- weight pre-pack into B-fragment layout (bf16) ---
  pack_w_kernel<<<(D_MODEL * D_HIDDEN + 255) / 256, 256, 0, stream>>>(W_pre, WpPre, D_MODEL, D_HIDDEN);
  pack_w_kernel<<<(D_RADIAL * D_HIDDEN + 255) / 256, 256, 0, stream>>>(W_filt, WpFilt, D_RADIAL, D_HIDDEN);
  pack_w_kernel<<<(D_HIDDEN * D_MODEL + 255) / 256, 256, 0, stream>>>(W_post, WpPost, D_HIDDEN, D_MODEL);
  pack_w_kernel<<<(D_MODEL * D_FF + 255) / 256, 256, 0, stream>>>(W_ff1, WpFF1, D_MODEL, D_FF);
  pack_w_kernel<<<(D_FF * D_MODEL + 255) / 256, 256, 0, stream>>>(W_ff2, WpFF2, D_FF, D_MODEL);

  // --- zero the scatter accumulator (must be re-done every call) ---
  zero_kernel<<<(N_NODES * D_HIDDEN + 255) / 256, 256, 0, stream>>>(agg, N_NODES * D_HIDDEN);

  // --- 1) LN1 -> bf16 ---
  ln_kernel<<<N_NODES, 256, 0, stream>>>(x, ln1_s, ln1_b, ln1bf);

  // --- 2) h_pre = ln1 @ W_pre + b_pre  (bf16 out for cheap gather) ---
  gemm_kernel<0, false><<<dim3(N_NODES / 16, D_HIDDEN / 256), 128, 0, stream>>>(
      ln1bf, WpPre, b_pre, nullptr, hpre, D_MODEL, D_HIDDEN);

  // --- 3) fused edge: filt WMMA + gather h_pre[src] + atomic scatter ---
  edge_kernel<<<N_EDGES / 16, 128, 0, stream>>>(
      radial, WpFilt, b_filt, hpre, src, dst, agg);

  // --- 4) x2 = silu(agg @ W_post + b_post) + x ---
  gemm_kernel<1, true><<<dim3(N_NODES / 16, D_MODEL / 256), 128, 0, stream>>>(
      agg, WpPost, b_post, x, x2, D_HIDDEN, D_MODEL);

  // --- 5) LN2 -> bf16 ---
  ln_kernel<<<N_NODES, 256, 0, stream>>>(x2, ln2_s, ln2_b, ln2bf);

  // --- 6) g = silu(ln2 @ W_ff1 + b_ff1) (bf16) ---
  gemm_kernel<2, false><<<dim3(N_NODES / 16, D_FF / 256), 128, 0, stream>>>(
      ln2bf, WpFF1, b_ff1, nullptr, gbf, D_MODEL, D_FF);

  // --- 7) out = g @ W_ff2 + b_ff2 + x2 ---
  gemm_kernel<3, false><<<dim3(N_NODES / 16, D_MODEL / 256), 128, 0, stream>>>(
      gbf, WpFF2, b_ff2, x2, d_out, D_FF, D_MODEL);
}
